// Sparsenet_81003083203115
// MI455X (gfx1250) — matrix operand order repaired and checked by reading, not compiled, not run on hardware
//
#include <hip/hip_runtime.h>
#include <hip/hip_bf16.h>

// ---------------------------------------------------------------------------
// SparseNet forward for MI455X (gfx1250, wave32).
// Conv / FC layers run as implicit GEMM on v_wmma_f32_16x16x32_bf16 with f32
// accumulation. Weights + per-layer activations pre-converted to bf16 by
// streaming kernels. Conv block tile M=32 x N=32 (4 waves), K step 64 for
// deep layers. Weight tiles are staged LDS-side with
// global_load_async_to_lds_b128 (ASYNCcnt) issued by wave 0 and overlapped
// with the A-tile gather. Total math ~0.7 TFLOP, streaming ~3 GB -> well
// under 1 ms against 23.3 TB/s HBM + WMMA pipes.
// ---------------------------------------------------------------------------

typedef __attribute__((ext_vector_type(16))) __bf16 v16bf;
typedef __attribute__((ext_vector_type(8)))  float  v8f;

union BF16x16 { v16bf v; unsigned short u[16]; };

__device__ __forceinline__ unsigned short f2bf(float f) {
    unsigned int u = __float_as_uint(f);
    unsigned int r = u + 0x7FFFu + ((u >> 16) & 1u);   // round-to-nearest-even
    return (unsigned short)(r >> 16);
}

__device__ __forceinline__ void async_wait0() {
#if defined(__has_builtin) && __has_builtin(__builtin_amdgcn_s_wait_asynccnt)
    __builtin_amdgcn_s_wait_asynccnt(0);
#else
    asm volatile("s_wait_asynccnt 0" ::: "memory");
#endif
}

__global__ void to_bf16(const float* __restrict__ x,
                        unsigned short* __restrict__ y, long n)
{
    const long i = (long)blockIdx.x * blockDim.x + threadIdx.x;
    if (i < n) y[i] = f2bf(x[i]);
}

// ---------------------------------------------------------------------------
// Implicit-GEMM 3D conv on bf16 inputs/weights:
//   out[vox, co] = (sum_tap sum_ci in[...]*w[...]) (+bias) (*mask) (+=out)
// in (B,X,Y,Z,Cin) bf16, w (kx,ky,kz,Cin,Cout) bf16, out (B,oX,oY,oZ,Cout) f32.
// Block = 128 threads (4 waves). Tile: M=32 voxels x N=32 out-channels;
// wave w -> (mSub = w&1, nSub = w>>1). K step 64 when Cin>=64.
// ---------------------------------------------------------------------------
__global__ void __launch_bounds__(128)
conv_wmma(const unsigned short* __restrict__ in,
          const unsigned short* __restrict__ w,
          const float* __restrict__ bias, const float* __restrict__ mask,
          float* __restrict__ out,
          int iX, int iY, int iZ, int Cin,
          int oX, int oY, int oZ, int Cout,
          int kX, int kY, int kZ,
          int sX, int sY, int sZ,
          int pX, int pY, int pZ, int accumulate)
{
    __shared__ unsigned short lA[32][72];   // 32 voxels x <=64 ch; rows 16B-aligned
    __shared__ unsigned short lB[64][40];   // <=64 ch x 32 cout; rows 16B-aligned

    const int tid  = threadIdx.x;
    const int wave = tid >> 5;
    const int lane = tid & 31;
    const int mSub = (wave & 1) << 4;
    const int nSub = (wave >> 1) << 4;
    const int nVox = 256 * oX * oY * oZ;
    const int voxBase = blockIdx.x * 32;
    const int coBase  = blockIdx.y * 32;
    const int nTap = kX * kY * kZ;
    const int spat = oX * oY * oZ;

    const int kStep = (Cin >= 64) ? 64 : 32;
    const bool vecA   = (Cin % 8) == 0;     // all layers except c0 (Cin=3)
    const bool asyncB = Cin >= 32;          // weight rows need no zero padding

    // fragment lane geometry (CDNA5 16-bit WMMA layouts)
    const int aM  = mSub + (lane & 15);
    const int aKb = (lane < 16) ? 0 : 8;
    const int bN  = nSub + (lane & 15);
    const int bKb = (lane < 16) ? 0 : 8;

    // ---- hoisted voxel decode for the vectorized A path -------------------
    // Thread stages groups i = tid, tid+128 of 8 channels; the voxel each
    // group belongs to is fixed across taps/chunks -> decode once.
    const int grpPerVox = kStep >> 3;       // 4 or 8
    const int nGrp = 32 * grpPerVox;        // 128 or 256
    int cb[2], cx[2], cy[2], cz[2];
    bool vok[2] = {false, false};
    for (int j = 0; j < 2; ++j) {
        const int i = tid + (j << 7);
        if (i >= nGrp) break;
        const int m = i / grpPerVox;
        const int vox = voxBase + m;
        if (vox < nVox) {
            const int b  = vox / spat;
            int r        = vox % spat;
            const int ox = r / (oY * oZ); r %= (oY * oZ);
            const int oy = r / oZ;
            const int oz = r % oZ;
            vok[j] = true;
            cb[j] = b;
            cx[j] = ox * sX - pX;
            cy[j] = oy * sY - pY;
            cz[j] = oz * sZ - pZ;
        }
    }

    v8f acc = {0.f, 0.f, 0.f, 0.f, 0.f, 0.f, 0.f, 0.f};

    for (int t = 0; t < nTap; ++t) {
        const int tz = t % kZ;
        const int ty = (t / kZ) % kY;
        const int tx = t / (kZ * kY);
        for (int kBase = 0; kBase < Cin; kBase += kStep) {
            // ---- issue async LDS copy of the B (weight) tile from wave 0 --
            if (asyncB && wave == 0) {
                const int nIss = kStep >> 3;        // 16B segments per lane
                for (int j2 = 0; j2 < nIss; ++j2) {
                    const int idx = (j2 << 5) + lane;
                    const int k   = idx >> 2;
                    const int seg = idx & 3;
                    const unsigned short* gp =
                        &w[((long)t * Cin + kBase + k) * Cout + coBase + (seg << 3)];
                    const unsigned long long ga = (unsigned long long)(uintptr_t)gp;
                    const unsigned int lp =
                        (unsigned int)(uintptr_t)&lB[k][seg << 3];
                    asm volatile("global_load_async_to_lds_b128 %0, %1, off"
                                 :: "v"(lp), "v"(ga) : "memory");
                }
            }
            __builtin_prefetch(&w[((long)t * Cin + kBase) * Cout + coBase], 0, 1);

            // ---- stage A tile (32 x kStep), overlapped with async B -------
            if (vecA) {
                for (int i = tid; i < nGrp; i += 128) {
                    const int j = (i - tid) >> 7;
                    const int k = (i % grpPerVox) << 3;
                    uint4 val = {0u, 0u, 0u, 0u};
                    if (vok[j]) {
                        const int ix = cx[j] + tx;
                        const int iy = cy[j] + ty;
                        const int iz = cz[j] + tz;
                        if (ix >= 0 && ix < iX && iy >= 0 && iy < iY &&
                            iz >= 0 && iz < iZ) {
                            const long ib =
                                ((long)(cb[j] * iX + ix) * iY + iy) * iZ + iz;
                            val = *(const uint4*)&in[ib * Cin + kBase + k];
                        }
                    }
                    const int m = i / grpPerVox;
                    *(uint4*)&lA[m][k] = val;
                }
            } else {
                for (int i = tid; i < 32 * kStep; i += 128) {
                    const int m = i / kStep, k = i % kStep;
                    const int vox = voxBase + m;
                    unsigned short val = 0;
                    if (vox < nVox && (kBase + k) < Cin) {
                        const int b  = vox / spat;
                        int r        = vox % spat;
                        const int ox = r / (oY * oZ); r %= (oY * oZ);
                        const int oy = r / oZ;
                        const int oz = r % oZ;
                        const int ix = ox * sX - pX + tx;
                        const int iy = oy * sY - pY + ty;
                        const int iz = oz * sZ - pZ + tz;
                        if (ix >= 0 && ix < iX && iy >= 0 && iy < iY &&
                            iz >= 0 && iz < iZ)
                            val = in[(((long)(b * iX + ix) * iY + iy) * iZ + iz) *
                                         Cin + kBase + k];
                    }
                    lA[m][k] = val;
                }
            }

            // ---- B tile: wait for async, or scalar fallback ---------------
            if (asyncB) {
                if (wave == 0) async_wait0();
            } else {
                for (int i = tid; i < kStep * 32; i += 128) {
                    const int k = i >> 5, co = i & 31;
                    unsigned short val = 0;
                    if ((kBase + k) < Cin)
                        val = w[((long)t * Cin + kBase + k) * Cout + coBase + co];
                    lB[k][co] = val;
                }
            }
            __syncthreads();

            // ---- chunk 0: K = kBase .. kBase+31 ---------------------------
            {
                BF16x16 fa, fb;
                #pragma unroll
                for (int e = 0; e < 16; ++e)
                    fa.u[e] = lA[aM][aKb + ((e < 8) ? e : e + 8)];
                #pragma unroll
                for (int e = 0; e < 16; ++e)
                    fb.u[e] = lB[bKb + ((e < 8) ? e : e + 8)][bN];
                acc = __builtin_amdgcn_wmma_f32_16x16x32_bf16(
                    false, fa.v, false, fb.v, (short)0, acc, false, false);
            }
            // ---- chunk 1: K = kBase+32 .. kBase+63 (deep layers) ----------
            if (kStep == 64) {
                BF16x16 fa, fb;
                #pragma unroll
                for (int e = 0; e < 16; ++e)
                    fa.u[e] = lA[aM][32 + aKb + ((e < 8) ? e : e + 8)];
                #pragma unroll
                for (int e = 0; e < 16; ++e)
                    fb.u[e] = lB[32 + bKb + ((e < 8) ? e : e + 8)][bN];
                acc = __builtin_amdgcn_wmma_f32_16x16x32_bf16(
                    false, fa.v, false, fb.v, (short)0, acc, false, false);
            }
            __syncthreads();
        }
    }

    // ---- write D tile (lane holds col N=lane&15, rows r + 8*(lane>=16)) ----
    const int co = coBase + nSub + (lane & 15);
    const int mRowBase = mSub + ((lane < 16) ? 0 : 8);
    #pragma unroll
    for (int r = 0; r < 8; ++r) {
        const int vox = voxBase + mRowBase + r;
        if (vox < nVox && co < Cout) {
            float v = acc[r];
            if (bias) v += bias[co];
            if (mask) v *= mask[vox];
            const long oi = (long)vox * Cout + co;
            out[oi] = accumulate ? (out[oi] + v) : v;
        }
    }
}

// ---------------------------------------------------------------------------
// Point-cloud scatter into dense grid + occupancy count (global atomics).
// ---------------------------------------------------------------------------
__global__ void scatter_kernel(const float* __restrict__ pc,
                               float* __restrict__ grid,
                               float* __restrict__ cnt, int npts)
{
    const int i = blockIdx.x * blockDim.x + threadIdx.x;
    if (i >= npts) return;
    const int x = (int)pc[i * 7 + 0];
    const int y = (int)pc[i * 7 + 1];
    const int z = (int)pc[i * 7 + 2];
    const int b = (int)pc[i * 7 + 3];
    const int vox = ((b * 10 + x) * 10 + y) * 40 + z;
    atomicAdd(&grid[vox * 3 + 0], pc[i * 7 + 4]);
    atomicAdd(&grid[vox * 3 + 1], pc[i * 7 + 5]);
    atomicAdd(&grid[vox * 3 + 2], pc[i * 7 + 6]);
    atomicAdd(&cnt[vox], 1.0f);
}

__global__ void mask_from_cnt(float* __restrict__ m, long n)
{
    const long i = (long)blockIdx.x * blockDim.x + threadIdx.x;
    if (i < n) m[i] = (m[i] > 0.f) ? 1.f : 0.f;
}

__global__ void zero_kernel(float* __restrict__ p, long n)
{
    const long i = (long)blockIdx.x * blockDim.x + threadIdx.x;
    if (i < n) p[i] = 0.f;
}

// ---------------------------------------------------------------------------
// Masked BN reduce: red[0..C)=sum(x*m), red[C..2C)=sum(x^2*m), red[2C]=sum(m)
// ---------------------------------------------------------------------------
__global__ void bn_reduce(const float* __restrict__ x,
                          const float* __restrict__ mask,
                          float* __restrict__ red, int nVox, int C)
{
    __shared__ float ls[1024];
    __shared__ float lq[1024];
    __shared__ float ln;
    for (int c = threadIdx.x; c < C; c += blockDim.x) { ls[c] = 0.f; lq[c] = 0.f; }
    if (threadIdx.x == 0) ln = 0.f;
    __syncthreads();

    const long total = (long)nVox * C;
    for (long e = (long)blockIdx.x * blockDim.x + threadIdx.x; e < total;
         e += (long)gridDim.x * blockDim.x) {
        const int  c   = (int)(e % C);
        const long vox = e / C;
        const float m  = mask ? mask[vox] : 1.f;
        const float xv = x[e];
        atomicAdd(&ls[c], xv * m);
        atomicAdd(&lq[c], xv * xv * m);
        if (c == 0) atomicAdd(&ln, m);
    }
    __syncthreads();
    for (int c = threadIdx.x; c < C; c += blockDim.x) {
        atomicAdd(&red[c], ls[c]);
        atomicAdd(&red[C + c], lq[c]);
    }
    if (threadIdx.x == 0) atomicAdd(&red[2 * C], ln);
}

__global__ void bn_apply(const float* __restrict__ x, float* __restrict__ y,
                         const float* __restrict__ red,
                         const float* __restrict__ g, const float* __restrict__ b,
                         const float* __restrict__ mask,
                         int nVox, int C, float eps, float leak)
{
    const long total = (long)nVox * C;
    const long e = (long)blockIdx.x * blockDim.x + threadIdx.x;
    if (e >= total) return;
    const int  c   = (int)(e % C);
    const long vox = e / C;
    const float n    = fmaxf(red[2 * C], 1.f);
    const float mean = red[c] / n;
    const float var  = red[C + c] / n - mean * mean;
    float v = (x[e] - mean) * rsqrtf(var + eps) * g[c] + b[c];
    v = (v > 0.f) ? v : leak * v;
    if (mask) v *= mask[vox];
    y[e] = v;
}

// ---------------------------------------------------------------------------
// Max-pool of mask (VALID windows).
// ---------------------------------------------------------------------------
__global__ void maskpool(const float* __restrict__ mi, float* __restrict__ mo,
                         int iX, int iY, int iZ,
                         int oX, int oY, int oZ,
                         int kX, int kY, int kZ,
                         int sX, int sY, int sZ)
{
    const int nOut = 256 * oX * oY * oZ;
    const int i = blockIdx.x * blockDim.x + threadIdx.x;
    if (i >= nOut) return;
    const int b  = i / (oX * oY * oZ);
    int r        = i % (oX * oY * oZ);
    const int ox = r / (oY * oZ); r %= (oY * oZ);
    const int oy = r / oZ;
    const int oz = r % oZ;
    float m = 0.f;
    for (int dx = 0; dx < kX; ++dx)
        for (int dy = 0; dy < kY; ++dy)
            for (int dz = 0; dz < kZ; ++dz) {
                const int ix = ox * sX + dx;
                const int iy = oy * sY + dy;
                const int iz = oz * sZ + dz;
                m = fmaxf(m, mi[((long)(b * iX + ix) * iY + iy) * iZ + iz]);
            }
    mo[i] = m;
}

// ---------------------------------------------------------------------------
// Final tiny FC: (256,1024) @ (1024,2) + bias -> d_out (512 floats).
// ---------------------------------------------------------------------------
__global__ void fc3_kernel(const float* __restrict__ h,
                           const float* __restrict__ w,
                           const float* __restrict__ b,
                           float* __restrict__ out)
{
    const int i = blockIdx.x * blockDim.x + threadIdx.x;
    if (i >= 512) return;
    const int bb = i >> 1, j = i & 1;
    float s = b[j];
    for (int k = 0; k < 1024; ++k)
        s += h[bb * 1024 + k] * w[k * 2 + j];
    out[i] = s;
}

// ---------------------------------------------------------------------------
extern "C" void kernel_launch(void* const* d_in, const int* in_sizes, int n_in,
                              void* d_out, int out_size, void* d_ws, size_t ws_size,
                              hipStream_t stream)
{
    (void)n_in; (void)out_size; (void)ws_size;
    auto in = [&](int i) { return (const float*)d_in[i]; };

    // Input index map (setup_inputs dict order):
    // 0 point_cloud; 1 c0; 2..7 blk1{g1,b1,w1,g2,b2,w2}; 8..13 blk2;
    // 14,15 bn3; 16 c1; 17..22 blk3; 23,24 bn5; 25 c2; 26..31 blk4;
    // 32,33 bn7; 34 c3; 35..40 blk5; 41,42 bn9; 43 c4;
    // 44 fc1; 45 fb1; 46,47 bnf1; 48 fc2; 49 fb2; 50,51 bnf2; 52 fc3; 53 fb3.

    float* F = (float*)d_ws;
    const size_t NBUF = 33554432;              // 32M floats (128 MB) each
    float* P0  = F;
    float* P1  = F + NBUF;
    float* P2  = F + 2 * NBUF;
    float* M0  = F + 3 * NBUF;                 // (256,10,10,40)  1,024,000
    float* M1  = M0 + 1048576;                 // (256,5,5,20)      128,000
    float* M2  = M1 + 131072;                  // (256,4,4,9)        36,864
    float* M3  = M2 + 65536;                   // (256,2,2,3)         3,072
    float* M4  = M3 + 4096;                    // (256,1,1,1)           256
    float* RED = M4 + 1024;                    // up to 2*1024+1
    unsigned short* XB = (unsigned short*)(RED + 4096);   // act bf16: 32M ushorts
    unsigned short* WB = XB + 33554432;                   // weights bf16

    auto zero = [&](float* p, long n) {
        zero_kernel<<<(unsigned)((n + 255) / 256), 256, 0, stream>>>(p, n);
    };
    auto cvt = [&](const float* src, unsigned short* dst, long n) {
        to_bf16<<<(unsigned)((n + 255) / 256), 256, 0, stream>>>(src, dst, n);
    };

    // ---- convert all conv/FC weights to bf16 once ----
    const int widx[] = {1, 4, 7, 10, 13, 16, 19, 22, 25, 28, 31, 34, 37, 40,
                        43, 44, 48};
    const unsigned short* wb[54];
    size_t woff = 0;
    for (int i = 0; i < 17; ++i) {
        const int idx = widx[i];
        const long n = in_sizes[idx];
        cvt(in(idx), WB + woff, n);
        wb[idx] = WB + woff;
        woff += (size_t)((n + 7) & ~7L);
    }

    auto conv = [&](const float* x, int widx_, const float* bias,
                    const float* mask, float* y,
                    int iX, int iY, int iZ, int Cin,
                    int oX, int oY, int oZ, int Cout,
                    int kX, int kY, int kZ, int sX, int sY, int sZ,
                    int pX, int pY, int pZ, int acc) {
        const long nIn = 256L * iX * iY * iZ * Cin;
        cvt(x, XB, nIn);                       // bf16 staging of activations
        const int nVox = 256 * oX * oY * oZ;
        dim3 g((nVox + 31) / 32, Cout / 32);
        conv_wmma<<<g, 128, 0, stream>>>(XB, wb[widx_], bias, mask, y,
                                         iX, iY, iZ, Cin, oX, oY, oZ, Cout,
                                         kX, kY, kZ, sX, sY, sZ, pX, pY, pZ, acc);
    };
    auto bn = [&](const float* x, float* y, const float* g, const float* b,
                  const float* mask, int nVox, int C, float eps, float leak) {
        zero(RED, 2 * C + 1);
        bn_reduce<<<512, 256, 0, stream>>>(x, mask, RED, nVox, C);
        const long tot = (long)nVox * C;
        bn_apply<<<(unsigned)((tot + 255) / 256), 256, 0, stream>>>(
            x, y, RED, g, b, mask, nVox, C, eps, leak);
    };
    auto blk = [&](float* X, float* T1, float* T2, const float* mask,
                   int sx, int sy, int sz, int C, int pi) {
        const int nVox = 256 * sx * sy * sz;
        bn(X, T1, in(pi + 0), in(pi + 1), mask, nVox, C, 1e-4f, 0.f);
        conv(T1, pi + 2, nullptr, mask, T2, sx, sy, sz, C, sx, sy, sz, C,
             3, 3, 3, 1, 1, 1, 1, 1, 1, 0);
        bn(T2, T2, in(pi + 3), in(pi + 4), mask, nVox, C, 1e-4f, 0.f);
        conv(T2, pi + 5, nullptr, mask, X, sx, sy, sz, C, sx, sy, sz, C,
             3, 3, 3, 1, 1, 1, 1, 1, 1, 1);           // residual accumulate
    };

    // ---- scatter to dense grid + mask ----
    zero(P0, 256L * 4000 * 3);
    zero(M0, 256L * 4000);
    scatter_kernel<<<(131072 + 255) / 256, 256, 0, stream>>>(in(0), P0, M0, 131072);
    mask_from_cnt<<<(1024000 + 255) / 256, 256, 0, stream>>>(M0, 1024000);

    // ---- c0: (B,10,10,40,3) -> 32 ch, SAME, *mask ----
    conv(P0, 1, nullptr, M0, P1, 10, 10, 40, 3, 10, 10, 40, 32,
         3, 3, 3, 1, 1, 1, 1, 1, 1, 0);

    // ---- blk1, blk2 @ 32 ch ----
    blk(P1, P0, P2, M0, 10, 10, 40, 32, 2);
    blk(P1, P0, P2, M0, 10, 10, 40, 32, 8);

    // ---- bn3, down c1 (2,2,2)/(2,2,2): 32 -> 64, grid (5,5,20) ----
    bn(P1, P0, in(14), in(15), M0, 256 * 4000, 32, 1e-4f, 0.f);
    maskpool<<<(256 * 500 + 255) / 256, 256, 0, stream>>>(
        M0, M1, 10, 10, 40, 5, 5, 20, 2, 2, 2, 2, 2, 2);
    conv(P0, 16, nullptr, M1, P2, 10, 10, 40, 32, 5, 5, 20, 64,
         2, 2, 2, 2, 2, 2, 0, 0, 0, 0);

    // ---- blk3 @ 64 ch ----
    blk(P2, P0, P1, M1, 5, 5, 20, 64, 17);

    // ---- bn5, down c2 (2,2,4)/(1,1,2): 64 -> 256, grid (4,4,9) ----
    bn(P2, P0, in(23), in(24), M1, 256 * 500, 64, 1e-4f, 0.f);
    maskpool<<<(256 * 144 + 255) / 256, 256, 0, stream>>>(
        M1, M2, 5, 5, 20, 4, 4, 9, 2, 2, 4, 1, 1, 2);
    conv(P0, 25, nullptr, M2, P1, 5, 5, 20, 64, 4, 4, 9, 256,
         2, 2, 4, 1, 1, 2, 0, 0, 0, 0);

    // ---- blk4 @ 256 ch ----
    blk(P1, P0, P2, M2, 4, 4, 9, 256, 26);

    // ---- bn7, down c3 (2,2,3)/(2,2,3): 256 -> 512, grid (2,2,3) ----
    bn(P1, P0, in(32), in(33), M2, 256 * 144, 256, 1e-4f, 0.f);
    maskpool<<<(256 * 12 + 255) / 256, 256, 0, stream>>>(
        M2, M3, 4, 4, 9, 2, 2, 3, 2, 2, 3, 2, 2, 3);
    conv(P0, 34, nullptr, M3, P2, 4, 4, 9, 256, 2, 2, 3, 512,
         2, 2, 3, 2, 2, 3, 0, 0, 0, 0);

    // ---- blk5 @ 512 ch ----
    blk(P2, P0, P1, M3, 2, 2, 3, 512, 35);

    // ---- bn9, down c4 (2,2,3)/(2,2,3): 512 -> 1024, grid (1,1,1) ----
    bn(P2, P0, in(41), in(42), M3, 256 * 12, 512, 1e-4f, 0.f);
    maskpool<<<1, 256, 0, stream>>>(M3, M4, 2, 2, 3, 1, 1, 1, 2, 2, 3, 2, 2, 3);
    conv(P0, 43, nullptr, M4, P1, 2, 2, 3, 512, 1, 1, 1, 1024,
         2, 2, 3, 2, 2, 3, 0, 0, 0, 0);

    // ---- FC head: fc1 + bn1d + leaky, fc2 + bn1d + leaky, fc3 ----
    conv(P1, 44, in(45), nullptr, P2, 1, 1, 1, 1024, 1, 1, 1, 1024,
         1, 1, 1, 1, 1, 1, 0, 0, 0, 0);
    bn(P2, P2, in(46), in(47), nullptr, 256, 1024, 1e-5f, 0.01f);
    conv(P2, 48, in(49), nullptr, P0, 1, 1, 1, 1024, 1, 1, 1, 1024,
         1, 1, 1, 1, 1, 1, 0, 0, 0, 0);
    bn(P0, P0, in(50), in(51), nullptr, 256, 1024, 1e-5f, 0.01f);
    fc3_kernel<<<2, 256, 0, stream>>>(P0, in(52), in(53), (float*)d_out);
}
